// Similariy_Net_58222576664919
// MI455X (gfx1250) — matrix operand order repaired and checked by reading, not compiled, not run on hardware
//
#include <hip/hip_runtime.h>
#include <hip/hip_bf16.h>

typedef __attribute__((ext_vector_type(16))) __bf16 bf16x16;
typedef __attribute__((ext_vector_type(8)))  __bf16 bf16x8;
typedef __attribute__((ext_vector_type(4)))  __bf16 bf16x4;
typedef __attribute__((ext_vector_type(2)))  __bf16 bf16x2;
typedef __attribute__((ext_vector_type(8)))  float  f32x8;

#define NSAMP 20000
#define KPEER 20
#define FDIM  64
#define HDIM  128
#define WAVES_PER_BLOCK 2

// ---- pre-transposed bf16 weight offsets in d_ws (elements). Wt is [Nout][K] row-major.
#define O_IRM1 0                      // 128x64
#define O_IRM2 (O_IRM1 + 128*64)      // 64x128
#define O_IRR  (O_IRM2 + 64*128)      // 128x64
#define O_IRC  (O_IRR  + 128*64)      // 128x128
#define O_PRM1 (O_IRC  + 128*128)     // 128x64
#define O_PRM2 (O_PRM1 + 128*64)      // 64x128
#define O_PRR  (O_PRM2 + 64*128)      // 128x64
#define O_PRC  (O_PRR  + 128*64)      // 128x128
#define O_SR1  (O_PRC  + 128*128)     // 128x128
#define O_SR2  (O_SR1  + 128*128)
#define O_SM1  (O_SR2  + 128*128)
#define O_SM2  (O_SM1  + 128*128)
#define WT_TOTAL (O_SM2 + 128*128)    // 147456 bf16 = 294912 bytes

struct KParams {
  const float* M;         // [NREC,64]
  const float* OBS;       // [NREC,128]
  const float* fea_corr;  // [64,64]
  const int*   obs_idx;   // [N]
  const int*   fea_idx;   // [N]
  const int*   peer_idx;  // [N,20]
  const float* bias[12];  // I1..4, P1..4, sr1, sr2, sm1, sm2
  const float* sc_w;      // [128,128] fp32 (untransposed)
  const float* sc_b;      // [128]
  const __bf16* wt;       // transposed bf16 weights in ws
  float* out_csim;        // [N,128]
  float* out_sim;         // [N,20]
};

struct alignas(16) WaveLds {
  __bf16 X [32*128];   // activation tile, row-major (ld 64 or 128)
  __bf16 Ci[32*128];   // saved c_i
  __bf16 PH[32*128];   // peer_hidden (bf16)
  __bf16 PM[32*64];    // peer_mask   (bf16)
  __bf16 SH[128];      // sample_hidden
  __bf16 SMK[64];      // sample_mask
  __bf16 AR[64];       // fea_corr[fea_idx]
  float  simv[32];
  float  esim[128];
};

// Branchless GELU (tanh form) on the CDNA5 hardware tanh when exposed.
__device__ __forceinline__ float gelu_fast(float x) {
  const float u = 0.7978845608028654f * x * (1.0f + 0.044715f * x * x);
#if __has_builtin(__builtin_amdgcn_tanhf)
  const float t = __builtin_amdgcn_tanhf(u);
#else
  const float e = __expf(2.0f * u);
  const float t = 1.0f - 2.0f * __builtin_amdgcn_rcpf(e + 1.0f);
#endif
  return 0.5f * x * (1.0f + t);
}

// Computes C^T = Wt * X^T:  A = Wt[NOUT][KDIM] (global bf16, rows = out cols),
// B = Xin rows (LDS row-major). C fragment: lane holds 8 CONSECUTIVE out cols of
// one X-row -> epilogue is one packed 16B ds_store_b128 per (mt, nt), and the
// bias vector directly initializes the WMMA C accumulator (added exactly once).
// All B fragments (entire Xin) preloaded before any store => in-place safe.
template<int KDIM, int LDI, int NOUT, int LDO, bool HASMUL, int MSTRIDE>
__device__ __forceinline__ void gemm_stage(const __bf16* Xin,
                                           const __bf16* __restrict__ Wt,
                                           const float*  __restrict__ bias,
                                           const __bf16* mulbuf,
                                           __bf16* Xout,
                                           int lane) {
  constexpr int KT = KDIM / 32;
  constexpr int MT = NOUT / 16;   // tiles over output columns (WMMA M dim)
  const int r  = lane & 15;
  const int kh = lane >> 4;

  __syncthreads();  // prior-phase LDS writes -> visible / no reordering

  // ---- preload all B fragments: lane = X-row (r within n-tile), contiguous K
  bf16x16 bfr[2 * KT];
#pragma unroll
  for (int nt = 0; nt < 2; ++nt) {
#pragma unroll
    for (int kt = 0; kt < KT; ++kt) {
      const __bf16* p = Xin + (nt * 16 + r) * LDI + kt * 32 + kh * 16;
      bf16x8 c0 = *(const bf16x8*)(p);
      bf16x8 c1 = *(const bf16x8*)(p + 8);
      bf16x16 b;
#pragma unroll
      for (int i = 0; i < 8; ++i) { b[i] = c0[i]; b[i + 8] = c1[i]; }
      bfr[nt * KT + kt] = b;
    }
  }

#pragma unroll
  for (int mt = 0; mt < MT; ++mt) {
    // 8 consecutive out cols for this lane: m = mt*16 + kh*8 + v
    const int mbase = mt * 16 + kh * 8;
    const float4 bl = *(const float4*)(bias + mbase);
    const float4 bh = *(const float4*)(bias + mbase + 4);
    const f32x8 binit = {bl.x, bl.y, bl.z, bl.w, bh.x, bh.y, bh.z, bh.w};
    f32x8 acc0 = binit;
    f32x8 acc1 = binit;
#pragma unroll
    for (int kt = 0; kt < KT; ++kt) {
      // A fragment: lane = out col (row of Wt), contiguous K chunk kh*16..kh*16+15
      const __bf16* wp = Wt + (mt * 16 + r) * KDIM + kt * 32 + kh * 16;
      bf16x8 a0 = *(const bf16x8*)(wp);
      bf16x8 a1 = *(const bf16x8*)(wp + 8);
      bf16x16 a;
#pragma unroll
      for (int i = 0; i < 8; ++i) { a[i] = a0[i]; a[i + 8] = a1[i]; }
      acc0 = __builtin_amdgcn_wmma_f32_16x16x32_bf16(false, a, false, bfr[kt],      (short)0, acc0, false, false);
      acc1 = __builtin_amdgcn_wmma_f32_16x16x32_bf16(false, a, false, bfr[KT + kt], (short)0, acc1, false, false);
    }
    bf16x8 g8;
    if (HASMUL && MSTRIDE == 0) g8 = *(const bf16x8*)(mulbuf + mbase);
    auto epi = [&](const f32x8& acc, int nt) {
      bf16x8 gp;
      if (HASMUL && MSTRIDE != 0) gp = *(const bf16x8*)(mulbuf + (nt * 16 + r) * MSTRIDE + mbase);
      bf16x8 outv;
#pragma unroll
      for (int v = 0; v < 8; ++v) {
        float val = gelu_fast(acc[v]);
        if (HASMUL) val *= (float)(MSTRIDE != 0 ? gp[v] : g8[v]);
        outv[v] = (__bf16)val;
      }
      *(bf16x8*)(Xout + (nt * 16 + r) * LDO + mbase) = outv;
    };
    epi(acc0, 0);
    epi(acc1, 1);
  }
  __syncthreads();
}

// softmax over F of (mask[f] * (f != feaI)) for one row; vectorized 8-wide.
__device__ __forceinline__ void softmax_row(const __bf16* maskrow, int feaI,
                                            __bf16* xrow) {
  bf16x8 mr[8];
#pragma unroll
  for (int c = 0; c < 8; ++c) mr[c] = *(const bf16x8*)(maskrow + c * 8);
  float mx = -1e30f;
#pragma unroll
  for (int c = 0; c < 8; ++c)
#pragma unroll
    for (int i = 0; i < 8; ++i) {
      float x = ((c * 8 + i) == feaI) ? 0.0f : (float)mr[c][i];
      mx = fmaxf(mx, x);
    }
  float sum = 0.0f;
#pragma unroll
  for (int c = 0; c < 8; ++c)
#pragma unroll
    for (int i = 0; i < 8; ++i) {
      float x = ((c * 8 + i) == feaI) ? 0.0f : (float)mr[c][i];
      sum += __expf(x - mx);
    }
  const float inv = __builtin_amdgcn_rcpf(sum);
#pragma unroll
  for (int c = 0; c < 8; ++c) {
    bf16x8 o;
#pragma unroll
    for (int i = 0; i < 8; ++i) {
      float x = ((c * 8 + i) == feaI) ? 0.0f : (float)mr[c][i];
      o[i] = (__bf16)(__expf(x - mx) * inv);
    }
    *(bf16x8*)(xrow + c * 8) = o;
  }
}

__global__ void __launch_bounds__(32 * WAVES_PER_BLOCK)
simnet_kernel(KParams kp) {
  __shared__ WaveLds smem[WAVES_PER_BLOCK];
  const int wave = threadIdx.x >> 5;
  const int lane = threadIdx.x & 31;
  const int n = blockIdx.x * WAVES_PER_BLOCK + wave;
  WaveLds& S = smem[wave];

  const int obsI = kp.obs_idx[n];
  const int feaI = kp.fea_idx[n];

  // ---- gather phase (L2-resident tables), vectorized ----
  {
    const float2 mv = *(const float2*)(kp.M + obsI * FDIM + lane * 2);
    bf16x2 m2; m2[0] = (__bf16)mv.x; m2[1] = (__bf16)mv.y;
    *(bf16x2*)(S.SMK + lane * 2) = m2;
    const float2 av = *(const float2*)(kp.fea_corr + feaI * FDIM + lane * 2);
    bf16x2 a2; a2[0] = (__bf16)av.x; a2[1] = (__bf16)av.y;
    *(bf16x2*)(S.AR + lane * 2) = a2;
    const float4 hv = *(const float4*)(kp.OBS + obsI * HDIM + lane * 4);
    bf16x4 h4; h4[0] = (__bf16)hv.x; h4[1] = (__bf16)hv.y; h4[2] = (__bf16)hv.z; h4[3] = (__bf16)hv.w;
    *(bf16x4*)(S.SH + lane * 4) = h4;
  }
#pragma unroll 4
  for (int k = 0; k < KPEER; ++k) {
    const int pi = kp.peer_idx[n * KPEER + k];
    const float2 mv = *(const float2*)(kp.M + pi * FDIM + lane * 2);
    bf16x2 m2; m2[0] = (__bf16)mv.x; m2[1] = (__bf16)mv.y;
    *(bf16x2*)(S.PM + k * FDIM + lane * 2) = m2;
    const float4 hv = *(const float4*)(kp.OBS + pi * HDIM + lane * 4);
    bf16x4 h4; h4[0] = (__bf16)hv.x; h4[1] = (__bf16)hv.y; h4[2] = (__bf16)hv.z; h4[3] = (__bf16)hv.w;
    *(bf16x4*)(S.PH + k * HDIM + lane * 4) = h4;
  }
  {
    bf16x2 z2 = {(__bf16)0.0f, (__bf16)0.0f};
    bf16x4 z4 = {(__bf16)0.0f, (__bf16)0.0f, (__bf16)0.0f, (__bf16)0.0f};
#pragma unroll
    for (int k = KPEER; k < 32; ++k) {   // zero padding rows
      *(bf16x2*)(S.PM + k * FDIM + lane * 2) = z2;
      *(bf16x4*)(S.PH + k * HDIM + lane * 4) = z4;
    }
  }
  __syncthreads();

  // ================= I attr-net =================
  softmax_row(&S.PM[lane * FDIM], feaI, &S.X[lane * FDIM]);   // lane == row
  gemm_stage< 64, 64,128,128,false,  0>(S.X, kp.wt + O_IRM1, kp.bias[0], nullptr, S.X, lane);
  gemm_stage<128,128, 64, 64,true,   0>(S.X, kp.wt + O_IRM2, kp.bias[1], S.AR,    S.X, lane);
  gemm_stage< 64, 64,128,128,true,   0>(S.X, kp.wt + O_IRR,  kp.bias[2], S.SH,    S.X, lane);
  gemm_stage<128,128,128,128,false,  0>(S.X, kp.wt + O_IRC,  kp.bias[3], nullptr, S.Ci, lane);

  // ================= P attr-net =================
  softmax_row(S.SMK, feaI, &S.X[lane * FDIM]);                // broadcast row
  gemm_stage< 64, 64,128,128,false,  0>(S.X, kp.wt + O_PRM1, kp.bias[4], nullptr, S.X, lane);
  gemm_stage<128,128, 64, 64,true,   0>(S.X, kp.wt + O_PRM2, kp.bias[5], S.AR,    S.X, lane);
  gemm_stage< 64, 64,128,128,true, 128>(S.X, kp.wt + O_PRR,  kp.bias[6], S.PH,    S.X, lane);
  gemm_stage<128,128,128,128,false,  0>(S.X, kp.wt + O_PRC,  kp.bias[7], nullptr, S.X, lane);  // c_t

  // ================= cosine(c_i, c_t) =================
  {
    const int r = lane;
    float num = 0.f, na = 0.f, nb = 0.f;
#pragma unroll
    for (int h8 = 0; h8 < HDIM / 8; ++h8) {
      bf16x8 av = *(const bf16x8*)(S.Ci + r * HDIM + h8 * 8);
      bf16x8 bv = *(const bf16x8*)(S.X  + r * HDIM + h8 * 8);
#pragma unroll
      for (int i = 0; i < 8; ++i) {
        float a = (float)av[i], b = (float)bv[i];
        num += a * b; na += a * a; nb += b * b;
      }
    }
    float s = num / (fmaxf(sqrtf(na), 1e-8f) * fmaxf(sqrtf(nb), 1e-8f));
    s = (r < KPEER) ? s : 0.0f;
    S.simv[r] = s;
    if (r < KPEER) kp.out_sim[n * KPEER + r] = s;
  }
  __syncthreads();

  // ================= m_q = [peer_mask | keep_mask(feaI)] =================
  {
    const int r = lane;
#pragma unroll
    for (int c = 0; c < 8; ++c)
      *(bf16x8*)(S.X + r * HDIM + c * 8) = *(const bf16x8*)(S.PM + r * FDIM + c * 8);
#pragma unroll
    for (int c = 0; c < 8; ++c) {
      bf16x8 o;
#pragma unroll
      for (int i = 0; i < 8; ++i) o[i] = (__bf16)(((c * 8 + i) == feaI) ? 1.0f : 0.0f);
      *(bf16x8*)(S.X + r * HDIM + FDIM + c * 8) = o;
    }
  }
  gemm_stage<128,128,128,128,false,  0>(S.X, kp.wt + O_SR1, kp.bias[8],  nullptr, S.X, lane);
  gemm_stage<128,128,128,128,true, 128>(S.X, kp.wt + O_SR2, kp.bias[9],  S.PH,    S.X, lane);  // r_t_j * peer_hidden
  gemm_stage<128,128,128,128,false,  0>(S.X, kp.wt + O_SM1, kp.bias[10], nullptr, S.X, lane);
  gemm_stage<128,128,128,128,false,  0>(S.X, kp.wt + O_SM2, kp.bias[11], nullptr, S.X, lane);  // h_query

  // ================= e_sim = sum_k sim[k] * h_query[k] =================
  {
    const int c = lane * 4;
    float e0 = 0.f, e1 = 0.f, e2 = 0.f, e3 = 0.f;
#pragma unroll 4
    for (int k = 0; k < KPEER; ++k) {
      const float s = S.simv[k];
      const bf16x4 hp = *(const bf16x4*)(S.X + k * HDIM + c);
      e0 += s * (float)hp[0]; e1 += s * (float)hp[1];
      e2 += s * (float)hp[2]; e3 += s * (float)hp[3];
    }
    S.esim[c + 0] = e0; S.esim[c + 1] = e1; S.esim[c + 2] = e2; S.esim[c + 3] = e3;
  }
  __syncthreads();

  // ================= c_sim = gelu(e_sim @ sc_w + sc_b) (fp32 matvec) =================
  {
    const int j = lane * 4;
    float a0 = kp.sc_b[j + 0], a1 = kp.sc_b[j + 1], a2 = kp.sc_b[j + 2], a3 = kp.sc_b[j + 3];
#pragma unroll 8
    for (int h = 0; h < HDIM; ++h) {
      const float e = S.esim[h];
      const float4 w = *(const float4*)(kp.sc_w + h * HDIM + j);
      a0 += e * w.x; a1 += e * w.y; a2 += e * w.z; a3 += e * w.w;
    }
    float* o = kp.out_csim + n * HDIM + j;
    o[0] = gelu_fast(a0); o[1] = gelu_fast(a1);
    o[2] = gelu_fast(a2); o[3] = gelu_fast(a3);
  }
}

// transpose fp32 [K][N] -> bf16 [N][K]
__global__ void wprep_kernel(const float* __restrict__ src, __bf16* __restrict__ dst,
                             int K, int N) {
  const int i = blockIdx.x * blockDim.x + threadIdx.x;
  if (i < K * N) {
    const int n = i / K, k = i % K;
    dst[i] = (__bf16)src[k * N + n];
  }
}

extern "C" void kernel_launch(void* const* d_in, const int* in_sizes, int n_in,
                              void* d_out, int out_size, void* d_ws, size_t ws_size,
                              hipStream_t stream) {
  (void)in_sizes; (void)n_in; (void)out_size; (void)ws_size;

  __bf16* wt = (__bf16*)d_ws;

  struct WtDesc { int in_idx; int off; int K; int N; };
  const WtDesc wd[12] = {
    { 6, O_IRM1,  64, 128}, { 8, O_IRM2, 128,  64}, {10, O_IRR,  64, 128}, {12, O_IRC, 128, 128},
    {14, O_PRM1,  64, 128}, {16, O_PRM2, 128,  64}, {18, O_PRR,  64, 128}, {20, O_PRC, 128, 128},
    {22, O_SR1, 128, 128}, {24, O_SR2, 128, 128}, {26, O_SM1, 128, 128}, {28, O_SM2, 128, 128},
  };
  for (int i = 0; i < 12; ++i) {
    const int total = wd[i].K * wd[i].N;
    wprep_kernel<<<(total + 255) / 256, 256, 0, stream>>>(
        (const float*)d_in[wd[i].in_idx], wt + wd[i].off, wd[i].K, wd[i].N);
  }

  KParams kp;
  kp.M        = (const float*)d_in[0];
  kp.OBS      = (const float*)d_in[1];
  kp.obs_idx  = (const int*)  d_in[2];
  kp.fea_idx  = (const int*)  d_in[3];
  kp.fea_corr = (const float*)d_in[4];
  kp.peer_idx = (const int*)  d_in[5];
  kp.bias[0]  = (const float*)d_in[7];
  kp.bias[1]  = (const float*)d_in[9];
  kp.bias[2]  = (const float*)d_in[11];
  kp.bias[3]  = (const float*)d_in[13];
  kp.bias[4]  = (const float*)d_in[15];
  kp.bias[5]  = (const float*)d_in[17];
  kp.bias[6]  = (const float*)d_in[19];
  kp.bias[7]  = (const float*)d_in[21];
  kp.bias[8]  = (const float*)d_in[23];
  kp.bias[9]  = (const float*)d_in[25];
  kp.bias[10] = (const float*)d_in[27];
  kp.bias[11] = (const float*)d_in[29];
  kp.sc_w     = (const float*)d_in[30];
  kp.sc_b     = (const float*)d_in[31];
  kp.wt       = wt;
  kp.out_csim = (float*)d_out;
  kp.out_sim  = (float*)d_out + (size_t)NSAMP * HDIM;

  simnet_kernel<<<NSAMP / WAVES_PER_BLOCK, 32 * WAVES_PER_BLOCK, 0, stream>>>(kp);
}
